// Attention_23441931502023
// MI455X (gfx1250) — matrix operand order repaired and checked by reading, not compiled, not run on hardware
//
#include <hip/hip_runtime.h>
#include <hip/hip_bf16.h>

// ---------------------------------------------------------------------------
// Fused attention for MI455X (gfx1250): bf16 WMMA + TDM async tensor staging.
// B=4, L=2048, D=1024, H=16, dk=dv=64.
// ---------------------------------------------------------------------------

typedef __bf16 bf16;
typedef __attribute__((ext_vector_type(16))) __bf16 v16bf;
typedef __attribute__((ext_vector_type(8)))  float  v8f;
typedef __attribute__((ext_vector_type(4))) unsigned int u32x4;
typedef __attribute__((ext_vector_type(8))) int         i32x8;
typedef __attribute__((ext_vector_type(4))) int         i32x4;

#define BATCH 4
#define SEQ   2048
#define DMODEL 1024
#define HEADS 16
#define DK    64
#define ROWS  (BATCH * SEQ)
#define NEG_BIG 1e10f

#if __has_builtin(__builtin_amdgcn_tensor_load_to_lds) && \
    __has_builtin(__builtin_amdgcn_s_wait_tensorcnt)
#define USE_TDM 1
#else
#define USE_TDM 0
#endif

__device__ inline v8f zero8() {
    v8f z;
#pragma unroll
    for (int i = 0; i < 8; ++i) z[i] = 0.0f;
    return z;
}

// A-fragment (16x32 bf16): 16 contiguous K-values starting at p0 (p1 unused).
__device__ inline v16bf load8x2(const bf16* p0, const bf16* p1) {
    union { uint4 u[2]; v16bf v; } t;
    t.u[0] = *(const uint4*)p0;
    t.u[1] = *(const uint4*)(p0 + 8);
    return t.v;
}

// B-fragment (32x16 bf16): 16 contiguous K-values for this lane's column.
__device__ inline v16bf load16c(const bf16* p) {
    union { uint4 u[2]; v16bf v; } t;
    t.u[0] = *(const uint4*)p;
    t.u[1] = *(const uint4*)(p + 8);
    return t.v;
}

__device__ inline v8f wmma_bf16(v16bf a, v16bf b, v8f c) {
    return __builtin_amdgcn_wmma_f32_16x16x32_bf16(
        false, a, false, b, (short)0, c, false, false);
}

#if USE_TDM
// Issue one TDM 2-D tile load: rows x rowlen bf16 elements, row stride
// `stride` elements, global -> LDS (row-major, packed). Wave-level op.
// D# packing per CDNA5 ISA 8.3/8.4: group0 = {count|flags, lds_addr,
// global_addr lo, global_addr hi | type=2}; group1 = {mask/data_size,
// tensor dims, tile dims, dim0 stride}; groups 2/3 unused (2-D tile).
__device__ inline void tdm_load_tile(const bf16* g, const bf16* l, int rows,
                                     int rowlen, int stride) {
    unsigned long long ga = (unsigned long long)(uintptr_t)g;
    unsigned int la = (unsigned int)(uintptr_t)l;  // low 32 bits = LDS offset
    u32x4 g0;
    g0[0] = 1u;                                        // count=1, user mode
    g0[1] = la;                                        // lds_addr
    g0[2] = (unsigned int)ga;                          // global_addr[31:0]
    g0[3] = (unsigned int)((ga >> 32) & 0x1FFFFFFu) |  // global_addr[56:32]
            (2u << 30);                                // type=2 (image)
    i32x8 g1;
    unsigned int ur = (unsigned int)rows, uw = (unsigned int)rowlen;
    g1[0] = (int)(1u << 16);                       // wg_mask=0, data_size=2B
    g1[1] = (int)((uw & 0xFFFFu) << 16);           // tensor_dim0 lo
    g1[2] = (int)(((uw >> 16) & 0xFFFFu) | ((ur & 0xFFFFu) << 16));
    g1[3] = (int)(((ur >> 16) & 0xFFFFu) | ((uw & 0xFFFFu) << 16)); // tile_dim0
    g1[4] = (int)(ur & 0xFFFFu);                   // tile_dim1, tile_dim2=0
    g1[5] = stride;                                // tensor_dim0_stride lo
    g1[6] = 0;
    g1[7] = 0;
    i32x4 z4 = {0, 0, 0, 0};
#if __has_include(<hip/amd_detail/amd_gfx1250_TDM.h>)
    i32x8 z8 = {0, 0, 0, 0, 0, 0, 0, 0};
    __builtin_amdgcn_tensor_load_to_lds(g0, g1, z4, z4, z8, 0);
#else
    __builtin_amdgcn_tensor_load_to_lds(g0, g1, z4, z4, 0);
#endif
}
#endif

// ---------------------------------------------------------------------------
// Kernel 1: fp32 -> bf16 conversion of q, k, v (d_in order is q, v, k!)
// ---------------------------------------------------------------------------
__global__ __launch_bounds__(256) void convert_qkv(
    const float* __restrict__ q, const float* __restrict__ v,
    const float* __restrict__ k,
    bf16* __restrict__ qbf, bf16* __restrict__ kbf, bf16* __restrict__ vbf) {
    const size_t N = (size_t)ROWS * DMODEL;
    size_t idx = (size_t)blockIdx.x * 256 + threadIdx.x;
    if (idx < N)          qbf[idx]         = (bf16)q[idx];
    else if (idx < 2 * N) kbf[idx - N]     = (bf16)k[idx - N];
    else if (idx < 3 * N) vbf[idx - 2 * N] = (bf16)v[idx - 2 * N];
}

// ---------------------------------------------------------------------------
// Kernel 2: transpose weights to [N][K] bf16 (B-fragments become contiguous)
// ---------------------------------------------------------------------------
__global__ __launch_bounds__(256) void transpose_w(
    const float* __restrict__ Wq, const float* __restrict__ Wk,
    const float* __restrict__ Wv,
    bf16* __restrict__ Wqt, bf16* __restrict__ Wkt, bf16* __restrict__ Wvt) {
    int idx = blockIdx.x * 256 + threadIdx.x;  // 0 .. 3*1048576-1
    int mat = idx >> 20;
    int r   = idx & 1048575;
    int k   = r >> 10;
    int n   = r & 1023;
    const float* W = (mat == 0) ? Wq : (mat == 1) ? Wk : Wv;
    bf16*       Wt = (mat == 0) ? Wqt : (mat == 1) ? Wkt : Wvt;
    Wt[(size_t)n * DMODEL + k] = (bf16)W[(size_t)k * DMODEL + n];
}

// ---------------------------------------------------------------------------
// Kernel 3: projection GEMMs, software-pipelined and unrolled x2 with
// ping-pong fragment register sets (no rotation moves): loads for slice
// k+32 / k+64 are in flight while WMMAs retire slice k.
// Q,K -> [B,H,L,64]; V -> transposed [B,H,64,L].
// ---------------------------------------------------------------------------
__global__ __launch_bounds__(256) void proj_gemm(
    const bf16* __restrict__ qbf, const bf16* __restrict__ kbf,
    const bf16* __restrict__ vbf,
    const bf16* __restrict__ Wqt, const bf16* __restrict__ Wkt,
    const bf16* __restrict__ Wvt,
    bf16* __restrict__ Qp, bf16* __restrict__ Kp, bf16* __restrict__ Vpt) {
    const int lane = threadIdx.x & 31;
    const int wid  = blockIdx.x * 8 + (threadIdx.x >> 5);  // 0..24575
    const int mat  = wid / 8192;                           // 0=Q 1=K 2=V
    const int rem  = wid % 8192;
    const int mtile = rem >> 4;
    const int n64   = rem & 15;

    const bf16* A  = (mat == 0) ? qbf : (mat == 1) ? kbf : vbf;
    const bf16* Wt = (mat == 0) ? Wqt : (mat == 1) ? Wkt : Wvt;

    const int col   = lane & 15;
    const int ka    = (lane < 16) ? 0 : 16;
    const int kb    = (lane < 16) ? 0 : 16;
    const int half8 = (lane >= 16) ? 8 : 0;

    const bf16* arow = A + (size_t)(mtile * 16 + col) * DMODEL + ka;
    const bf16* wrow[4];
#pragma unroll
    for (int c = 0; c < 4; ++c)
        wrow[c] = Wt + (size_t)(n64 * 64 + c * 16 + col) * DMODEL + kb;

    v8f acc[4];
#pragma unroll
    for (int c = 0; c < 4; ++c) acc[c] = zero8();

    // ping-pong fragment sets: set 0 holds slice k0, set 1 holds k0+32
    v16bf a0 = load8x2(arow, arow + 16);
    v16bf b0[4];
#pragma unroll
    for (int c = 0; c < 4; ++c) b0[c] = load16c(wrow[c]);

    for (int k0 = 0; k0 < DMODEL; k0 += 64) {
        v16bf a1 = load8x2(arow + k0 + 32, arow + k0 + 48);
        v16bf b1[4];
#pragma unroll
        for (int c = 0; c < 4; ++c) b1[c] = load16c(wrow[c] + k0 + 32);
#pragma unroll
        for (int c = 0; c < 4; ++c) acc[c] = wmma_bf16(a0, b0[c], acc[c]);
        if (k0 + 64 < DMODEL) {
            a0 = load8x2(arow + k0 + 64, arow + k0 + 80);
#pragma unroll
            for (int c = 0; c < 4; ++c) b0[c] = load16c(wrow[c] + k0 + 64);
        }
#pragma unroll
        for (int c = 0; c < 4; ++c) acc[c] = wmma_bf16(a1, b1[c], acc[c]);
    }

#pragma unroll
    for (int c = 0; c < 4; ++c) {
#pragma unroll
        for (int r = 0; r < 8; ++r) {
            int m = mtile * 16 + r + half8;
            int n = n64 * 64 + c * 16 + col;
            int b = m >> 11, l = m & (SEQ - 1);
            int h = n >> 6,  d = n & (DK - 1);
            bf16 val = (bf16)acc[c][r];
            if (mat == 0)
                Qp[(((size_t)(b * HEADS + h)) * SEQ + l) * DK + d] = val;
            else if (mat == 1)
                Kp[(((size_t)(b * HEADS + h)) * SEQ + l) * DK + d] = val;
            else
                Vpt[(((size_t)(b * HEADS + h)) * DK + d) * SEQ + l] = val;
        }
    }
}

// ---------------------------------------------------------------------------
// Kernel 4: causal flash attention with double-buffered TDM staging of K/V
// tiles into LDS. grid = (L/64, B*H), block = 128 (4 waves, 16 q-rows each).
// All waves run a block-uniform tile count; compute is causally guarded.
// ---------------------------------------------------------------------------
__global__ __launch_bounds__(128) void flash_attn(
    const bf16* __restrict__ Qp, const bf16* __restrict__ Kp,
    const bf16* __restrict__ Vpt,
    const float* __restrict__ v_mask, const float* __restrict__ q_mask,
    float* __restrict__ out) {
    __shared__ __align__(16) bf16 Kbuf[2][32 * DK];   // 2 x 4KB: 32 keys x 64
    __shared__ __align__(16) bf16 Vbuf[2][DK * 32];   // 2 x 4KB: 64 dv x 32 keys
    __shared__ __align__(16) bf16 Pbuf[4][16 * 32];   // per-wave P staging

    const int lane  = threadIdx.x & 31;
    const int wave  = threadIdx.x >> 5;
    const int bh    = blockIdx.y;
    const int b     = bh >> 4;
    const int h     = bh & 15;
    const int qbase = blockIdx.x * 64 + wave * 16;

    const bf16* Qh = Qp  + (size_t)bh * SEQ * DK;
    const bf16* Kh = Kp  + (size_t)bh * SEQ * DK;
    const bf16* Vh = Vpt + (size_t)bh * DK * SEQ;

    const int col   = lane & 15;
    const int ka    = (lane < 16) ? 0 : 16;
    const int kb    = (lane < 16) ? 0 : 16;
    const int half8 = (lane >= 16) ? 8 : 0;

    // Q A-fragments (16 rows x 64 depth), kept in registers.
    const bf16* qptr = Qh + (size_t)(qbase + col) * DK;
    v16bf qa0 = load8x2(qptr + ka,      qptr + ka + 16);
    v16bf qa1 = load8x2(qptr + 32 + ka, qptr + 48 + ka);

    v8f acc[4];
#pragma unroll
    for (int c = 0; c < 4; ++c) acc[c] = zero8();
    float mst[8], lst[8];
#pragma unroll
    for (int r = 0; r < 8; ++r) { mst[r] = -1e30f; lst[r] = 0.0f; }

    const int myT = qbase / 32 + 1;        // this wave's causal tile count
    const int T   = 2 * blockIdx.x + 2;    // block-uniform max tile count

#if !USE_TDM
    // synchronous fallback: cooperative copy of a 32-key K/V tile pair
    auto coop_copy = [&](int k0, int buf) {
        for (int c = threadIdx.x; c < 256; c += 128) {
            int key = c >> 3, off = (c & 7) * 8;
            *(uint4*)&Kbuf[buf][key * DK + off] =
                *(const uint4*)(Kh + (size_t)(k0 + key) * DK + off);
        }
        for (int c = threadIdx.x; c < 256; c += 128) {
            int dv = c >> 2, off = (c & 3) * 8;
            *(uint4*)&Vbuf[buf][dv * 32 + off] =
                *(const uint4*)(Vh + (size_t)dv * SEQ + k0 + off);
        }
    };
#endif

    // prologue: stage tile 0 into buffer 0
#if USE_TDM
    if (wave == 0) tdm_load_tile(Kh, Kbuf[0], 32, DK, DK);
    if (wave == 1) tdm_load_tile(Vh, Vbuf[0], DK, 32, SEQ);
#else
    coop_copy(0, 0);
#endif

    for (int jt = 0; jt < T; ++jt) {
        const int cur = jt & 1;
        const int k0  = jt * 32;

        // stage next tile into the other buffer, keep its DMA in flight
        if (jt + 1 < T) {
            const int k0n = k0 + 32;
#if USE_TDM
            if (wave == 0)
                tdm_load_tile(Kh + (size_t)k0n * DK, Kbuf[1 - cur], 32, DK, DK);
            if (wave == 1)
                tdm_load_tile(Vh + k0n, Vbuf[1 - cur], DK, 32, SEQ);
            if (wave < 2) __builtin_amdgcn_s_wait_tensorcnt(1);
#else
            coop_copy(k0n, 1 - cur);
#endif
        } else {
#if USE_TDM
            if (wave < 2) __builtin_amdgcn_s_wait_tensorcnt(0);
#endif
        }
        __syncthreads();  // tile `jt` visible to all waves

        if (jt < myT) {
            // S = Q K^T from LDS K tile: two 16x16 C-frags, 2 WMMAs each
            const bf16* kr0 = &Kbuf[cur][col * DK + kb];
            const bf16* kr1 = &Kbuf[cur][(16 + col) * DK + kb];
            v8f c0 = zero8(), c1 = zero8();
            c0 = wmma_bf16(qa0, load16c(kr0), c0);
            c0 = wmma_bf16(qa1, load16c(kr0 + 32), c0);
            c1 = wmma_bf16(qa0, load16c(kr1), c1);
            c1 = wmma_bf16(qa1, load16c(kr1 + 32), c1);

            const float pen0 = (1.0f - v_mask[b * SEQ + k0 + col]) * NEG_BIG;
            const float pen1 =
                (1.0f - v_mask[b * SEQ + k0 + 16 + col]) * NEG_BIG;

            float alpha[8];
#pragma unroll
            for (int r = 0; r < 8; ++r) {
                const int qrow = qbase + r + half8;
                float s0 = c0[r] * 0.125f - pen0 -
                           ((k0 + col)      > qrow ? NEG_BIG : 0.0f);
                float s1 = c1[r] * 0.125f - pen1 -
                           ((k0 + 16 + col) > qrow ? NEG_BIG : 0.0f);
                float rm = fmaxf(s0, s1);
#pragma unroll
                for (int off = 1; off < 16; off <<= 1)
                    rm = fmaxf(rm, __shfl_xor(rm, off, 32));
                const float mnew = fmaxf(mst[r], rm);
                const float a    = __expf(mst[r] - mnew);
                const float p0   = __expf(s0 - mnew);
                const float p1   = __expf(s1 - mnew);
                float rs = p0 + p1;
#pragma unroll
                for (int off = 1; off < 16; off <<= 1)
                    rs += __shfl_xor(rs, off, 32);
                lst[r] = lst[r] * a + rs;
                mst[r] = mnew;
                alpha[r] = a;
                Pbuf[wave][(r + half8) * 32 + col]      = (bf16)p0;
                Pbuf[wave][(r + half8) * 32 + 16 + col] = (bf16)p1;
            }

#pragma unroll
            for (int c = 0; c < 4; ++c)
#pragma unroll
                for (int r = 0; r < 8; ++r) acc[c][r] *= alpha[r];

            // per-wave LDS visibility for the P re-layout (C-frag -> A-frag)
            asm volatile("s_wait_dscnt 0" ::: "memory");
            const bf16* pp = &Pbuf[wave][col * 32];
            v16bf pa = load8x2(pp + ka, pp + ka + 16);

            // O += P @ V from LDS V tile (row = dv, contiguous over keys)
#pragma unroll
            for (int c = 0; c < 4; ++c) {
                const bf16* vp = &Vbuf[cur][(c * 16 + col) * 32 + kb];
                acc[c] = wmma_bf16(pa, load16c(vp), acc[c]);
            }
        }
        __syncthreads();  // all waves done reading tile before its buffer
                          // is overwritten by the next iteration's prefetch
    }

    // epilogue: divide by row sum, apply q_mask, store [B, L, H*64] fp32
#pragma unroll
    for (int r = 0; r < 8; ++r) {
        const int qrow = qbase + r + half8;
        const float f = (1.0f / lst[r]) * q_mask[b * SEQ + qrow];
#pragma unroll
        for (int c = 0; c < 4; ++c) {
            out[(size_t)(b * SEQ + qrow) * (HEADS * DK) + h * DK + c * 16 +
                col] = acc[c][r] * f;
        }
    }
}

// ---------------------------------------------------------------------------
// Host launcher
// ---------------------------------------------------------------------------
extern "C" void kernel_launch(void* const* d_in, const int* in_sizes, int n_in,
                              void* d_out, int out_size, void* d_ws,
                              size_t ws_size, hipStream_t stream) {
    // setup_inputs order: q, v, k, v_mask, q_mask, Wq, Wk, Wv
    const float* q      = (const float*)d_in[0];
    const float* v      = (const float*)d_in[1];
    const float* k      = (const float*)d_in[2];
    const float* v_mask = (const float*)d_in[3];
    const float* q_mask = (const float*)d_in[4];
    const float* Wq     = (const float*)d_in[5];
    const float* Wk     = (const float*)d_in[6];
    const float* Wv     = (const float*)d_in[7];

    const size_t NX = (size_t)ROWS * DMODEL;    // 8388608 elements
    const size_t NW = (size_t)DMODEL * DMODEL;  // 1048576 elements

    bf16* ws  = (bf16*)d_ws;
    bf16* qbf = ws;
    bf16* kbf = qbf + NX;
    bf16* vbf = kbf + NX;
    bf16* Wqt = vbf + NX;
    bf16* Wkt = Wqt + NW;
    bf16* Wvt = Wkt + NW;
    bf16* Qp  = Wvt + NW;
    bf16* Kp  = Qp + NX;
    bf16* Vpt = Kp + NX;
    // total: 6*NX + 3*NW bf16 = ~102 MB of d_ws

    convert_qkv<<<(3 * NX) / 256, 256, 0, stream>>>(q, v, k, qbf, kbf, vbf);
    transpose_w<<<(3 * NW) / 256, 256, 0, stream>>>(Wq, Wk, Wv, Wqt, Wkt, Wvt);
    proj_gemm<<<3072, 256, 0, stream>>>(qbf, kbf, vbf, Wqt, Wkt, Wvt, Qp, Kp,
                                        Vpt);
    flash_attn<<<dim3(SEQ / 64, BATCH * HEADS), 128, 0, stream>>>(
        Qp, Kp, Vpt, v_mask, q_mask, (float*)d_out);
}
